// Attention_53944789237811
// MI455X (gfx1250) — compile-verified, matrix operand order checked by reading, not run on hardware
//
#include <hip/hip_runtime.h>

// ---------------------------------------------------------------------------
// MI455X (gfx1250) attention pipeline.
//   keys    = LN(Y @ K)        queries = LN(X @ Q)
//   alpha   = softmax(queries @ keys^T / 1024)
//   out     = alpha @ Y
// All contractions: v_wmma_f32_16x16x32_bf16 with f32 accumulation.
// B operands double-buffered in LDS via global_load_async_to_lds_b128
// (ASYNCcnt) so async DMA of tile k+1 overlaps the WMMAs of tile k.
// ---------------------------------------------------------------------------

typedef __attribute__((ext_vector_type(16))) __bf16 v16bf;
typedef __attribute__((ext_vector_type(8)))  float  v8f;

#define HID    1024
#define SEQ    2048
#define NBATCH 8
#define NROWS  (NBATCH * SEQ)      // 16384
#define LN_EPS 1e-5f

// round-to-nearest-even float -> bf16 (raw u16)
static __device__ __forceinline__ unsigned short f2bf(float x) {
  unsigned u = __builtin_bit_cast(unsigned, x);
  unsigned r = u + 0x7FFFu + ((u >> 16) & 1u);
  return (unsigned short)(r >> 16);
}

union FragU { v16bf v; uint4 q[2]; };

// 16x32 bf16 fragment from row-major global memory at column k0.
// CDNA5 16-bit A layout: lanes 0-15 hold row M=lane, K in {0..7,16..23};
// lanes 16-31 hold row M=lane-16, K in {8..15,24..31}. Two b128 loads/lane.
static __device__ __forceinline__ v16bf load_frag16x32(
    const unsigned short* __restrict__ p, int ld, int k0, int lane) {
  const int half = (lane >> 4) & 1;
  const int r    = lane & 15;
  const unsigned short* base = p + (size_t)r * ld + (k0 + half * 8);
  FragU f;
  f.q[0] = *reinterpret_cast<const uint4*>(base);
  f.q[1] = *reinterpret_cast<const uint4*>(base + 16);
  return f.v;
}

// Same fragment pattern read from an LDS tile of layout [n][32] bf16.
static __device__ __forceinline__ v16bf lds_frag(
    const unsigned short* __restrict__ sm, int n0, int lane) {
  const int half = (lane >> 4) & 1;
  const int r    = lane & 15;
  const unsigned short* base = sm + (size_t)(n0 + r) * 32 + half * 8;
  FragU f;
  f.q[0] = *reinterpret_cast<const uint4*>(base);   // ds_load_b128
  f.q[1] = *reinterpret_cast<const uint4*>(base + 16);
  return f.v;
}

static __device__ __forceinline__ v8f wmma_bf16(v16bf a, v16bf b, v8f c) {
  return __builtin_amdgcn_wmma_f32_16x16x32_bf16(false, a, false, b,
                                                 (short)0, c, false, false);
}

static __device__ __forceinline__ void wait_async0() {
  asm volatile("s_wait_asynccnt 0" ::: "memory");
}

// Async-copy one B tile (Nrows x 32 bf16, row stride ld elems in global,
// packed [n][32] in LDS) into LDS at byte offset lds_base.  Every thread
// moves 16 x 16B chunks; chunk c covers row n=c>>2, quarter part=c&3.
static __device__ __forceinline__ void stage_tile(
    unsigned lds_base, const unsigned short* __restrict__ g, int ld,
    int tid, int nthreads) {
#pragma unroll
  for (int i = 0; i < 16; ++i) {
    const int c    = tid + i * nthreads;
    const int n    = c >> 2;
    const int part = c & 3;
    const unsigned long long gp =
        (unsigned long long)(uintptr_t)(g + (size_t)n * ld + part * 8);
    const unsigned lo = lds_base + (unsigned)c * 16u;
    asm volatile("global_load_async_to_lds_b128 %0, %1, off"
                 :: "v"(lo), "v"(gp) : "memory");
  }
}

// Double-buffered pipelined mainloop: acc[t] += A[16,K] * B[K, col0+t*16..+16]
// A read from global (register-pipelined); B staged async into LDS.
// Block stages B rows [0, nthreads*64/16) = all rows its waves consume.
static __device__ __forceinline__ void gemm_pipeline(
    const unsigned short* __restrict__ Arow, int ldA,
    const unsigned short* __restrict__ B0,  int ldB,
    int Kdim, unsigned short* sm, unsigned tileBytes,
    int tid, int nthreads, int lane, int col0, v8f acc[8]) {
  const unsigned sbase = (unsigned)(unsigned long long)(uintptr_t)sm;
  const int nIter = Kdim / 32;
  const int mr = lane & 15;

  stage_tile(sbase, B0, ldB, tid, nthreads);
  wait_async0();
  __syncthreads();

  v16bf a = load_frag16x32(Arow, ldA, 0, lane);

  for (int it = 0; it < nIter; ++it) {
    const unsigned curOff = (it & 1) ? tileBytes : 0u;
    const unsigned nxtOff = (it & 1) ? 0u : tileBytes;
    const bool more = (it + 1 < nIter);

    // issue async DMA for tile it+1 (overlaps this iteration's compute)
    if (more)
      stage_tile(sbase + nxtOff, B0 + (size_t)(it + 1) * 32, ldB, tid, nthreads);

    // register-pipeline next A fragment
    v16bf an = a;
    if (more) {
      an = load_frag16x32(Arow, ldA, (it + 1) * 32, lane);
      __builtin_prefetch(Arow + (size_t)mr * ldA + (it + 1) * 32 + 32, 0, 0);
    }

    // batch the 8 B-fragment ds loads, then the 8 WMMAs
    const unsigned short* smc = sm + (curOff >> 1);
    v16bf bf[8];
#pragma unroll
    for (int t = 0; t < 8; ++t) bf[t] = lds_frag(smc, col0 + t * 16, lane);
#pragma unroll
    for (int t = 0; t < 8; ++t) acc[t] = wmma_bf16(a, bf[t], acc[t]);

    a = an;
    wait_async0();     // own copies for tile it+1 done (overlapped compute)
    __syncthreads();   // block-wide: tile it reads done, tile it+1 visible
  }
}

// ---------------------------------------------------------------------------
// Kernel 1: X,Y fp32 -> bf16 row-major; also write Y^T (per batch) in bf16.
// ---------------------------------------------------------------------------
__global__ void attn_conv_xy(const float* __restrict__ X,
                             const float* __restrict__ Y,
                             unsigned short* __restrict__ Xb,
                             unsigned short* __restrict__ Yb,
                             unsigned short* __restrict__ YT) {
  size_t idx = (size_t)blockIdx.x * blockDim.x + threadIdx.x;  // < 16777216
  float x = X[idx];
  float y = Y[idx];
  Xb[idx] = f2bf(x);
  unsigned short yb = f2bf(y);
  Yb[idx] = yb;
  int f = (int)(idx & (HID - 1));
  int s = (int)((idx >> 10) & (SEQ - 1));
  int b = (int)(idx >> 21);
  YT[((size_t)b * HID + f) * SEQ + s] = yb;   // Y^T[b][f][s]
}

// ---------------------------------------------------------------------------
// Kernel 2: K,Q fp32 [F,H] -> transposed bf16 [H,F]  (B-operand layout)
// ---------------------------------------------------------------------------
__global__ void attn_conv_w(const float* __restrict__ K,
                            const float* __restrict__ Q,
                            unsigned short* __restrict__ KT,
                            unsigned short* __restrict__ QT) {
  size_t idx = (size_t)blockIdx.x * blockDim.x + threadIdx.x;  // < 1048576
  int h = (int)(idx & (HID - 1));
  int f = (int)(idx >> 10);
  size_t t = (size_t)h * HID + f;
  KT[t] = f2bf(K[idx]);
  QT[t] = f2bf(Q[idx]);
}

// ---------------------------------------------------------------------------
// Kernel 3: fused GEMM + LayerNorm.  256 thr (8 waves): 16 rows x 1024 cols.
// Dynamic LDS: 2 x 64KB B-tile buffers.
// ---------------------------------------------------------------------------
__global__ void attn_gemm_ln(const unsigned short* __restrict__ A,
                             const unsigned short* __restrict__ WT,
                             const float* __restrict__ gamma,
                             const float* __restrict__ beta,
                             unsigned short* __restrict__ out) {
  extern __shared__ unsigned short smem[];          // 2 * 1024*32 bf16
  const int tid  = threadIdx.x;
  const int lane = tid & 31;
  const int wave = tid >> 5;            // 0..7
  const int half = (lane >> 4) & 1;
  const int mr   = lane & 15;
  const int row0 = blockIdx.x * 16;
  const int col0 = wave * 128;

  v8f acc[8];
#pragma unroll
  for (int t = 0; t < 8; ++t)
#pragma unroll
    for (int r = 0; r < 8; ++r) acc[t][r] = 0.0f;

  gemm_pipeline(A + (size_t)row0 * HID, HID, WT, HID, HID,
                smem, 1024u * 64u, tid, 256, lane, col0, acc);

  __shared__ float s_sum[16];
  __shared__ float s_sq[16];
  if (tid < 16) { s_sum[tid] = 0.0f; s_sq[tid] = 0.0f; }
  __syncthreads();

#pragma unroll
  for (int r = 0; r < 8; ++r) {
    float ps = 0.0f, pq = 0.0f;
#pragma unroll
    for (int t = 0; t < 8; ++t) { float v = acc[t][r]; ps += v; pq += v * v; }
#pragma unroll
    for (int m = 1; m <= 8; m <<= 1) {      // reduce over the 16-lane half
      ps += __shfl_xor(ps, m, 32);
      pq += __shfl_xor(pq, m, 32);
    }
    if (mr == 0) {
      int row = r + 8 * half;
      atomicAdd(&s_sum[row], ps);           // ds_add_f32
      atomicAdd(&s_sq[row], pq);
    }
  }
  __syncthreads();

  const float inv_h = 1.0f / (float)HID;
#pragma unroll
  for (int r = 0; r < 8; ++r) {
    const int row  = r + 8 * half;
    const float mu = s_sum[row] * inv_h;
    float var      = s_sq[row] * inv_h - mu * mu;
    const float rs = rsqrtf(var + LN_EPS);
#pragma unroll
    for (int t = 0; t < 8; ++t) {
      const int n  = col0 + t * 16 + mr;
      float v = (acc[t][r] - mu) * rs * gamma[n] + beta[n];
      out[(size_t)(row0 + row) * HID + n] = f2bf(v);
    }
  }
}

// ---------------------------------------------------------------------------
// Kernel 4: fused scores + softmax.  512 thr (16 waves): 16 q rows x 2048 keys.
// Dynamic LDS: 2 x 128KB key-tile buffers.
// ---------------------------------------------------------------------------
__global__ void attn_scores_softmax(const unsigned short* __restrict__ Qb,
                                    const unsigned short* __restrict__ Kb,
                                    unsigned short* __restrict__ alpha) {
  extern __shared__ unsigned short smem[];          // 2 * 2048*32 bf16
  const int tid  = threadIdx.x;
  const int lane = tid & 31;
  const int wave = tid >> 5;            // 0..15
  const int half = (lane >> 4) & 1;
  const int mr   = lane & 15;
  const int b    = blockIdx.x >> 7;     // /128
  const int qt   = blockIdx.x & 127;
  const int row0 = qt * 16;
  const int col0 = wave * 128;

  const unsigned short* Qrow = Qb + ((size_t)b * SEQ + row0) * HID;
  const unsigned short* Krow = Kb + (size_t)b * SEQ * HID;

  v8f acc[8];
#pragma unroll
  for (int t = 0; t < 8; ++t)
#pragma unroll
    for (int r = 0; r < 8; ++r) acc[t][r] = 0.0f;

  gemm_pipeline(Qrow, HID, Krow, HID, HID,
                smem, 2048u * 64u, tid, 512, lane, col0, acc);

  const float sc = 1.0f / (float)HID;
#pragma unroll
  for (int t = 0; t < 8; ++t)
#pragma unroll
    for (int r = 0; r < 8; ++r) acc[t][r] *= sc;

  __shared__ float s_red[256];          // [wave][row]
  __shared__ float s_rowmax[16];
  __shared__ float s_rowsum[16];

  // ---- row max ----
#pragma unroll
  for (int r = 0; r < 8; ++r) {
    float m = -3.0e38f;
#pragma unroll
    for (int t = 0; t < 8; ++t) m = fmaxf(m, acc[t][r]);
#pragma unroll
    for (int msk = 1; msk <= 8; msk <<= 1) m = fmaxf(m, __shfl_xor(m, msk, 32));
    if (mr == 0) s_red[wave * 16 + r + 8 * half] = m;
  }
  __syncthreads();
  if (tid < 16) {
    float m = -3.0e38f;
    for (int w = 0; w < 16; ++w) m = fmaxf(m, s_red[w * 16 + tid]);
    s_rowmax[tid] = m;
  }
  __syncthreads();

  // ---- exp + row sum ----
#pragma unroll
  for (int r = 0; r < 8; ++r) {
    const int row = r + 8 * half;
    const float rm = s_rowmax[row];
    float s = 0.0f;
#pragma unroll
    for (int t = 0; t < 8; ++t) {
      float e = __expf(acc[t][r] - rm);
      acc[t][r] = e;
      s += e;
    }
#pragma unroll
    for (int msk = 1; msk <= 8; msk <<= 1) s += __shfl_xor(s, msk, 32);
    if (mr == 0) s_red[wave * 16 + row] = s;
  }
  __syncthreads();
  if (tid < 16) {
    float s = 0.0f;
    for (int w = 0; w < 16; ++w) s += s_red[w * 16 + tid];
    s_rowsum[tid] = 1.0f / s;
  }
  __syncthreads();

  // ---- normalize + store bf16 alpha ----
#pragma unroll
  for (int r = 0; r < 8; ++r) {
    const int row = r + 8 * half;
    const float inv = s_rowsum[row];
#pragma unroll
    for (int t = 0; t < 8; ++t) {
      const int n = col0 + t * 16 + mr;
      alpha[((size_t)b * SEQ + row0 + row) * SEQ + n] = f2bf(acc[t][r] * inv);
    }
  }
}

// ---------------------------------------------------------------------------
// Kernel 5: out = alpha @ Y   (per batch [2048x2048] x [2048x1024], fp32 out)
// 256 thr (8 waves): 16 q rows x 1024 f cols; K-loop over 2048.
// Dynamic LDS: 2 x 64KB YT-tile buffers.
// ---------------------------------------------------------------------------
__global__ void attn_pv_gemm(const unsigned short* __restrict__ alpha,
                             const unsigned short* __restrict__ YT,
                             float* __restrict__ out) {
  extern __shared__ unsigned short smem[];          // 2 * 1024*32 bf16
  const int tid  = threadIdx.x;
  const int lane = tid & 31;
  const int wave = tid >> 5;            // 0..7
  const int half = (lane >> 4) & 1;
  const int mr   = lane & 15;
  const int b    = blockIdx.x >> 7;
  const int qt   = blockIdx.x & 127;
  const int row0 = qt * 16;
  const int col0 = wave * 128;

  const unsigned short* Arow = alpha + ((size_t)b * SEQ + row0) * SEQ;
  const unsigned short* Brow = YT + (size_t)b * HID * SEQ;

  v8f acc[8];
#pragma unroll
  for (int t = 0; t < 8; ++t)
#pragma unroll
    for (int r = 0; r < 8; ++r) acc[t][r] = 0.0f;

  gemm_pipeline(Arow, SEQ, Brow, SEQ, SEQ,
                smem, 1024u * 64u, tid, 256, lane, col0, acc);

#pragma unroll
  for (int r = 0; r < 8; ++r) {
    const int row = r + 8 * half;
#pragma unroll
    for (int t = 0; t < 8; ++t) {
      const int n = col0 + t * 16 + mr;
      out[((size_t)b * SEQ + row0 + row) * HID + n] = acc[t][r];
    }
  }
}

// ---------------------------------------------------------------------------
// host launcher
// ---------------------------------------------------------------------------
extern "C" void kernel_launch(void* const* d_in, const int* in_sizes, int n_in,
                              void* d_out, int out_size, void* d_ws, size_t ws_size,
                              hipStream_t stream) {
  const float* X  = (const float*)d_in[0];
  const float* Y  = (const float*)d_in[1];
  const float* K  = (const float*)d_in[2];
  const float* Q  = (const float*)d_in[3];
  const float* g1 = (const float*)d_in[4];
  const float* b1 = (const float*)d_in[5];
  const float* g2 = (const float*)d_in[6];
  const float* b2 = (const float*)d_in[7];
  float* out = (float*)d_out;

  // workspace layout (bytes); total ~228 MB
  char* w = (char*)d_ws;
  const size_t SZ_XY = (size_t)NROWS * HID * 2;        // 32 MB each
  const size_t SZ_W  = (size_t)HID * HID * 2;          // 2 MB each
  unsigned short* Xb    = (unsigned short*)(w);
  unsigned short* Yb    = (unsigned short*)(w + SZ_XY);
  unsigned short* YT    = (unsigned short*)(w + 2 * SZ_XY);
  unsigned short* KT    = (unsigned short*)(w + 3 * SZ_XY);
  unsigned short* QT    = (unsigned short*)(w + 3 * SZ_XY + SZ_W);
  unsigned short* Kb    = (unsigned short*)(w + 3 * SZ_XY + 2 * SZ_W);
  unsigned short* Qb    = (unsigned short*)(w + 4 * SZ_XY + 2 * SZ_W);
  unsigned short* alpha = (unsigned short*)(w + 5 * SZ_XY + 2 * SZ_W); // 64 MB

  const size_t LDS_1K = 2u * 1024u * 64u;   // 128 KB (double-buffered 64KB tiles)
  const size_t LDS_2K = 2u * 2048u * 64u;   // 256 KB

  // 1) convert X,Y -> bf16 (+ Y^T), K,Q -> transposed bf16
  attn_conv_xy<<<(NROWS * HID) / 256, 256, 0, stream>>>(X, Y, Xb, Yb, YT);
  attn_conv_w<<<(HID * HID) / 256, 256, 0, stream>>>(K, Q, KT, QT);

  // 2) keys = LN(Y @ K), queries = LN(X @ Q)   [fused GEMM+LN, bf16 out]
  attn_gemm_ln<<<NROWS / 16, 256, LDS_1K, stream>>>(Yb, KT, g1, b1, Kb);
  attn_gemm_ln<<<NROWS / 16, 256, LDS_1K, stream>>>(Xb, QT, g2, b2, Qb);

  // 3) alpha = softmax(queries @ keys^T / H)   [fused, bf16 out]
  attn_scores_softmax<<<NBATCH * (SEQ / 16), 512, LDS_2K, stream>>>(Qb, Kb, alpha);

  // 4) out = alpha @ Y                          [fp32 out]
  attn_pv_gemm<<<NBATCH * (SEQ / 16), 256, LDS_1K, stream>>>(alpha, YT, out);
}
